// LCATSSR_19653770347083
// MI455X (gfx1250) — matrix operand order repaired and checked by reading, not compiled, not run on hardware
//
#include <hip/hip_runtime.h>
#include <math.h>

// ---------------------------------------------------------------------------
// MI455X (gfx1250) implementation of GroupSR (stereo transformer, 6 blocks).
// GEMM work -> V_WMMA_F32_16X16X32_F16 (wave32, f32 accum).
// Conv activations staged to LDS via GLOBAL_LOAD_ASYNC_TO_LDS_B128, then
// transposed/converted in LDS so all fragments are 16B vector loads.
// ---------------------------------------------------------------------------

typedef __attribute__((ext_vector_type(16))) _Float16 v16h;
typedef __attribute__((ext_vector_type(8)))  _Float16 v8h;
typedef __attribute__((ext_vector_type(8)))  float    v8f;

#define BDIM 2
#define CDIM 64
#define HDIM 128
#define WDIM 256
#define HWSZ 32768           // HDIM*WDIM
#define NPIX 65536           // BDIM*HWSZ

#define WMMA_F32_F16(a, b, c) \
  __builtin_amdgcn_wmma_f32_16x16x32_f16(false, (a), false, (b), (short)0, (c), false, false)

// Assemble a 16-wide f16 A/B fragment from two 8-wide halves.
__device__ inline v16h frag_join(v8h lo, v8h hi) {
  v16h r;
#pragma unroll
  for (int j = 0; j < 8; ++j) { r[j] = lo[j]; r[j + 8] = hi[j]; }
  return r;
}

// ===========================================================================
// Pack f32 weights -> f16 (contiguous [Cout][64]).
// ===========================================================================
__global__ __launch_bounds__(256) void k_packw(
    const float* __restrict__ Wf, _Float16* __restrict__ Wh, int n)
{
  int i = blockIdx.x * 256 + threadIdx.x;
  if (i < n) Wh[i] = (_Float16)Wf[i];
}

// ===========================================================================
// 1x1 conv (Cin fixed = 64) as WMMA GEMM.
//   grid = NPIX/128 blocks, 256 threads (8 waves).
//   Stage X[64ch][128px] f32 via global_load_async_to_lds_b128 (32KB),
//   transpose+convert to f16 Xt[128px][72] in LDS (16B-aligned rows), then
//   waves sweep (co-tile, px-tile) pairs; all fragments are 16B loads.
// ===========================================================================
#define XT_STRIDE 72          // f16 elems per row: 144B = 9*16B (aligned, banks spread)

template<bool HASB, bool HASR>
__global__ __launch_bounds__(256) void k_conv1x1_lds(
    const float* __restrict__ X, const _Float16* __restrict__ Wh,
    const float* __restrict__ bias, const float* __restrict__ res,
    float* __restrict__ Y, int Cout)
{
  // single LDS object -> async LDS offsets are relative to 0
  __shared__ __attribute__((aligned(16))) char ldsmem[64 * 128 * 4 +
                                                      128 * XT_STRIDE * 2];
  float    (*Xs)[128]       = (float(*)[128])ldsmem;
  _Float16 (*Xt)[XT_STRIDE] = (_Float16(*)[XT_STRIDE])(ldsmem + 64 * 128 * 4);

  const int wv   = threadIdx.x >> 5;
  const int lane = threadIdx.x & 31;
  const int fn   = lane & 15;
  const int fh   = lane >> 4;
  const int p0   = blockIdx.x * 128;     // 128 pixels per block
  const int b    = p0 >> 15;
  const int w0   = p0 & (HWSZ - 1);
  const float* Xb = X + (size_t)b * CDIM * HWSZ + w0;

  // ---- async stage: 64 rows x 512B; each wave moves 8 rows, 16B per lane.
  for (int r = wv; r < 64; r += 8) {
    unsigned lds = (unsigned)(r * 512 + lane * 16);
    unsigned long long ga =
        (unsigned long long)(const void*)(Xb + (size_t)r * HWSZ + lane * 4);
    asm volatile("global_load_async_to_lds_b128 %0, %1, off"
                 :: "v"(lds), "v"(ga) : "memory");
  }
  asm volatile("s_wait_asynccnt 0" ::: "memory");
  __syncthreads();

  // ---- transpose + f32->f16 convert (px fastest: conflict-free reads).
  for (int e = threadIdx.x; e < 8192; e += 256) {
    int px = e & 127, k = e >> 7;
    Xt[px][k] = (_Float16)Xs[k][px];
  }
  __syncthreads();

  const int ntiles = (Cout >> 4) * 8;    // co tiles * px tiles
  for (int t = wv; t < ntiles; t += 8) {
    const int px0 = (t & 7) * 16;
    const int co0 = (t >> 3) * 16;
    v8f acc = {};
    const _Float16* wrow = Wh + (size_t)(co0 + fn) * 64;
    const _Float16* xrow = &Xt[px0 + fn][0];
#pragma unroll
    for (int k0 = 0; k0 < 64; k0 += 32) {
      v16h af = frag_join(*(const v8h*)(wrow + k0 + 8 * fh),
                          *(const v8h*)(wrow + k0 + 16 + 8 * fh));
      v16h bf = frag_join(*(const v8h*)(xrow + k0 + 8 * fh),
                          *(const v8h*)(xrow + k0 + 16 + 8 * fh));
      acc = WMMA_F32_F16(af, bf, acc);
    }
    // ---- epilogue: batched bias/res, unconditional stores.
    const size_t obase = ((size_t)b * Cout + co0) * HWSZ + w0 + px0 + fn;
    float bv[8], rv[8];
    if (HASB) {
#pragma unroll
      for (int r = 0; r < 8; ++r) bv[r] = bias[co0 + r + 8 * fh];
    }
    if (HASR) {
#pragma unroll
      for (int r = 0; r < 8; ++r)
        rv[r] = res[obase + (size_t)(r + 8 * fh) * HWSZ];
    }
#pragma unroll
    for (int r = 0; r < 8; ++r) {
      float v = acc[r];
      if (HASB) v += bv[r];
      if (HASR) v += rv[r];
      Y[obase + (size_t)(r + 8 * fh) * HWSZ] = v;
    }
  }
}

// ===========================================================================
// Generic WMMA attention, SEQ = 256, compile-time CQ/CV.
//   Qp,Kp: [N][256][CQ] (row-major, c contiguous)
//   Vt:    [N][CV][256] (transposed -> contiguous B fragments)
//   Yp:    [N][256][CV]
// block = 128 (4 waves); wave owns a 16-row tile; 16x256 f16 score tile in
// LDS; scores rewritten as exp() so P-fragments are ds_read_b128s; 1/rowsum
// folded at the store.
// ===========================================================================
template<int CQ, int CV>
__global__ __launch_bounds__(128) void k_attn(
    const _Float16* __restrict__ Qp, const _Float16* __restrict__ Kp,
    const _Float16* __restrict__ Vt, _Float16* __restrict__ Yp, float scale)
{
  __shared__ _Float16 Sbuf[4][16 * 256];
  __shared__ float rinv_s[4][16];

  const int g    = blockIdx.x;
  const int wv   = threadIdx.x >> 5;
  const int lane = threadIdx.x & 31;
  const int fn   = lane & 15;
  const int fh   = lane >> 4;

  const _Float16* Qg = Qp + (size_t)g * 256 * CQ;
  const _Float16* Kg = Kp + (size_t)g * 256 * CQ;
  const _Float16* Vg = Vt + (size_t)g * CV * 256;
  _Float16*       Yg = Yp + (size_t)g * 256 * CV;
  _Float16*       S  = Sbuf[wv];

  for (int rt = wv; rt < 16; rt += 4) {
    const int r0 = rt * 16;
    // ---- scores: S[16][256] = scale * Q_tile K^T
    const _Float16* qrow = Qg + (size_t)(r0 + fn) * CQ;
    for (int st = 0; st < 16; ++st) {
      const _Float16* krow = Kg + (size_t)(st * 16 + fn) * CQ;
      v8f acc = {};
#pragma unroll
      for (int k0 = 0; k0 < CQ; k0 += 32) {
        // A fragment (rows of Q): lo always valid; hi valid iff k0+32 <= CQ.
        v8h alo = *(const v8h*)(qrow + k0 + 8 * fh);
        v8h ahi = {};
        if (k0 + 32 <= CQ) ahi = *(const v8h*)(qrow + k0 + 16 + 8 * fh);
        v16h af = frag_join(alo, ahi);
        // B fragment (rows of K as columns): same layout as A (k0+8*fh, k0+16+8*fh).
        v8h blo = {}, bhi = {};
        if (k0 + 32 <= CQ) {
          blo = *(const v8h*)(krow + k0 + 8 * fh);
          bhi = *(const v8h*)(krow + k0 + 16 + 8 * fh);
        } else {                       // only low half of K-range valid
          blo = *(const v8h*)(krow + k0 + 8 * fh);
        }
        v16h bf = frag_join(blo, bhi);
        acc = WMMA_F32_F16(af, bf, acc);
      }
#pragma unroll
      for (int r = 0; r < 8; ++r)
        S[(r + 8 * fh) * 256 + st * 16 + fn] = (_Float16)(acc[r] * scale);
    }
    // ---- softmax: stats + rewrite row as exp(s - max) (lanes 0..15).
    if (lane < 16) {
      _Float16* row = S + lane * 256;
      float mx = -1e30f;
      for (int s = 0; s < 256; ++s) mx = fmaxf(mx, (float)row[s]);
      float sum = 0.f;
      for (int s = 0; s < 256; ++s) {
        float e = __expf((float)row[s] - mx);
        sum += e;
        row[s] = (_Float16)e;
      }
      rinv_s[wv][lane] = 1.f / sum;
    }
    // ---- Y_tile = P V   (P fragments from LDS, V fragments contiguous)
    const _Float16* prow = S + fn * 256;
#pragma unroll
    for (int nt = 0; nt < CV / 16; ++nt) {
      const _Float16* vrow = Vg + (size_t)(nt * 16 + fn) * 256;
      v8f acc = {};
#pragma unroll
      for (int k0 = 0; k0 < 256; k0 += 32) {
        v16h af = frag_join(*(const v8h*)(prow + k0 + 8 * fh),
                            *(const v8h*)(prow + k0 + 16 + 8 * fh));
        v16h bf = frag_join(*(const v8h*)(vrow + k0 + 8 * fh),
                            *(const v8h*)(vrow + k0 + 16 + 8 * fh));
        acc = WMMA_F32_F16(af, bf, acc);
      }
#pragma unroll
      for (int r = 0; r < 8; ++r) {
        int m = r + 8 * fh;
        Yg[(size_t)(r0 + m) * CV + nt * 16 + fn] =
            (_Float16)(acc[r] * rinv_s[wv][m]);
      }
    }
  }
}

// ===========================================================================
// LayerNorm over channels (C=64) per pixel.  grid = NPIX/256.
// ===========================================================================
__global__ __launch_bounds__(256) void k_ln2d(
    const float* __restrict__ X, const float* __restrict__ g,
    const float* __restrict__ bt, float* __restrict__ Y)
{
  int p = blockIdx.x * 256 + threadIdx.x;
  int b = p >> 15, off = p & (HWSZ - 1);
  const float* xp = X + (size_t)b * CDIM * HWSZ + off;
  float* yp = Y + (size_t)b * CDIM * HWSZ + off;
  float mu = 0.f, m2 = 0.f;
  for (int c = 0; c < CDIM; ++c) {
    float v = xp[(size_t)c * HWSZ];
    mu += v; m2 += v * v;
  }
  mu *= (1.f / CDIM);
  float var = m2 * (1.f / CDIM) - mu * mu;
  float inv = rsqrtf(var + 1e-6f);
  for (int c = 0; c < CDIM; ++c)
    yp[(size_t)c * HWSZ] = g[c] * ((xp[(size_t)c * HWSZ] - mu) * inv) + bt[c];
}

// ===========================================================================
// Depthwise 3x3, pad 1.  grid = (HWSZ/256, C, B).
// ===========================================================================
__global__ __launch_bounds__(256) void k_dw3x3(
    const float* __restrict__ X, const float* __restrict__ Wt,
    const float* __restrict__ bias, float* __restrict__ Y, int C)
{
  int i = blockIdx.x * 256 + threadIdx.x;
  int x = i & (WDIM - 1), y = i >> 8;
  int c = blockIdx.y, b = blockIdx.z;
  const float* wp = Wt + c * 9;
  const float* xp = X + ((size_t)b * C + c) * HWSZ;
  float acc = bias ? bias[c] : 0.f;
#pragma unroll
  for (int dy = -1; dy <= 1; ++dy) {
    int yy = y + dy;
    if (yy < 0 || yy >= HDIM) continue;
#pragma unroll
    for (int dx = -1; dx <= 1; ++dx) {
      int xx = x + dx;
      if (xx < 0 || xx >= WDIM) continue;
      acc += wp[(dy + 1) * 3 + (dx + 1)] * xp[yy * WDIM + xx];
    }
  }
  Y[((size_t)b * C + c) * HWSZ + i] = acc;
}

// ===========================================================================
// mean over H,W per (b,c); optional second addend.  grid = B*C, block = 256.
// ===========================================================================
__global__ __launch_bounds__(256) void k_mean_hw(
    const float* __restrict__ X1, const float* __restrict__ X2,
    float* __restrict__ S)
{
  __shared__ float red[256];
  int bc = blockIdx.x;
  const float* p1 = X1 + (size_t)bc * HWSZ;
  const float* p2 = X2 ? X2 + (size_t)bc * HWSZ : nullptr;
  float s = 0.f;
  for (int i = threadIdx.x; i < HWSZ; i += 256) {
    float v = p1[i];
    if (p2) v += p2[i];
    s += v;
  }
  red[threadIdx.x] = s;
  __syncthreads();
  for (int st = 128; st > 0; st >>= 1) {
    if (threadIdx.x < st) red[threadIdx.x] += red[threadIdx.x + st];
    __syncthreads();
  }
  if (threadIdx.x == 0) S[bc] = red[0] * (1.f / HWSZ);
}

// ===========================================================================
// Squeeze-excite MLP: gate = sigmoid(W2 * lrelu(W1 * s)).
// ===========================================================================
__global__ __launch_bounds__(64) void k_se(
    const float* __restrict__ S, const float* __restrict__ W1,
    const float* __restrict__ W2, float* __restrict__ Gate)
{
  __shared__ float sh_s[64];
  __shared__ float sh_h[16];
  int b = blockIdx.x, t = threadIdx.x;
  sh_s[t] = S[b * 64 + t];
  __syncthreads();
  if (t < 16) {
    float a = 0.f;
    for (int c = 0; c < 64; ++c) a += W1[t * 64 + c] * sh_s[c];
    sh_h[t] = (a > 0.f) ? a : 0.2f * a;
  }
  __syncthreads();
  float o = 0.f;
  for (int h = 0; h < 16; ++h) o += W2[t * 16 + h] * sh_h[h];
  Gate[b * 64 + t] = 1.f / (1.f + __expf(-o));
}

// mode 0: Y = g*X1.   mode 1: Y = X1*g + X2*(1-g).   grid=(HWSZ/256,C,B).
__global__ __launch_bounds__(256) void k_gate(
    const float* __restrict__ X1, const float* __restrict__ X2,
    const float* __restrict__ Gate, float* __restrict__ Y, int C, int mode)
{
  int i = blockIdx.x * 256 + threadIdx.x;
  int c = blockIdx.y, b = blockIdx.z;
  size_t o = ((size_t)b * C + c) * HWSZ + i;
  float g = Gate[b * C + c];
  Y[o] = (mode == 0) ? X1[o] * g : X1[o] * g + X2[o] * (1.f - g);
}

// copy channels [32,64) (the v2 half of a concat).  grid=(HWSZ/256,32,B).
__global__ __launch_bounds__(256) void k_copy_hi(
    const float* __restrict__ S, float* __restrict__ D)
{
  int i = blockIdx.x * 256 + threadIdx.x;
  int c = 32 + blockIdx.y, b = blockIdx.z;
  size_t o = ((size_t)b * CDIM + c) * HWSZ + i;
  D[o] = S[o];
}

// ===========================================================================
// Window pack/unpack.  Q/K: [g][256][16] row-major.  V: [g][16][256] (transp).
// g = ((b*8 + wy)*16 + wx)*2 + hd.
// ===========================================================================
__global__ __launch_bounds__(256) void k_pack_win(
    const float* __restrict__ X, _Float16* __restrict__ P, int c0, int sh)
{
  size_t i = (size_t)blockIdx.x * 256 + threadIdx.x;
  int c = i & 15, t = (i >> 4) & 255;
  int g = i >> 12;
  int hd = g & 1, wx = (g >> 1) & 15, wy = (g >> 5) & 7, b = g >> 8;
  int y = (wy * 16 + (t >> 4) + sh) & (HDIM - 1);
  int x = (wx * 16 + (t & 15) + sh) & (WDIM - 1);
  int ch = c0 + hd * 16 + c;
  P[i] = (_Float16)X[(((size_t)b * CDIM + ch) * HDIM + y) * WDIM + x];
}

__global__ __launch_bounds__(256) void k_pack_win_vt(
    const float* __restrict__ X, _Float16* __restrict__ P, int sh)
{
  size_t i = (size_t)blockIdx.x * 256 + threadIdx.x;  // [g][c][t]
  int t = i & 255, c = (i >> 8) & 15;
  int g = i >> 12;
  int hd = g & 1, wx = (g >> 1) & 15, wy = (g >> 5) & 7, b = g >> 8;
  int y = (wy * 16 + (t >> 4) + sh) & (HDIM - 1);
  int x = (wx * 16 + (t & 15) + sh) & (WDIM - 1);
  int ch = hd * 16 + c;                                // v1 = vs[:, :32]
  P[i] = (_Float16)X[(((size_t)b * CDIM + ch) * HDIM + y) * WDIM + x];
}

__global__ __launch_bounds__(256) void k_unpack_win(
    const _Float16* __restrict__ P, float* __restrict__ Y, int sh)
{
  size_t i = (size_t)blockIdx.x * 256 + threadIdx.x;
  int c = i & 15, t = (i >> 4) & 255;
  int g = i >> 12;
  int hd = g & 1, wx = (g >> 1) & 15, wy = (g >> 5) & 7, b = g >> 8;
  int y = (wy * 16 + (t >> 4) + sh) & (HDIM - 1);
  int x = (wx * 16 + (t & 15) + sh) & (WDIM - 1);
  int ch = hd * 16 + c;
  Y[(((size_t)b * CDIM + ch) * HDIM + y) * WDIM + x] = (float)P[i];
}

// ===========================================================================
// Cross-attn row pack/unpack.  Q/K: [n][256][Cc].  V: [n][64][256] (transp).
// grid = B*H, block = 256 (one x per thread).
// ===========================================================================
__global__ __launch_bounds__(256) void k_pack_rows(
    const float* __restrict__ X, _Float16* __restrict__ P, int Cc)
{
  int n = blockIdx.x;
  int b = n >> 7, h = n & (HDIM - 1);
  int x = threadIdx.x;
  _Float16* pp = P + ((size_t)n * WDIM + x) * Cc;
  for (int c = 0; c < Cc; ++c)
    pp[c] = (_Float16)X[(((size_t)b * Cc + c) * HDIM + h) * WDIM + x];
}

__global__ __launch_bounds__(256) void k_pack_rows_vt(
    const float* __restrict__ X, _Float16* __restrict__ P)
{
  int n = blockIdx.x;
  int b = n >> 7, h = n & (HDIM - 1);
  int x = threadIdx.x;
  for (int c = 0; c < CDIM; ++c)
    P[((size_t)n * CDIM + c) * WDIM + x] =
        (_Float16)X[(((size_t)b * CDIM + c) * HDIM + h) * WDIM + x];
}

__global__ __launch_bounds__(256) void k_unpack_rows(
    const _Float16* __restrict__ P, float* __restrict__ Y)
{
  int n = blockIdx.x;
  int b = n >> 7, h = n & (HDIM - 1);
  int x = threadIdx.x;
  const _Float16* pp = P + ((size_t)n * WDIM + x) * CDIM;
  for (int c = 0; c < CDIM; ++c)
    Y[(((size_t)b * CDIM + c) * HDIM + h) * WDIM + x] = (float)pp[c];
}

// ===========================================================================
// MLP GLU: Y[b,c] = gelu(E[b,c]) * E[b,c+64].  grid = (HWSZ/256, 64, B).
// ===========================================================================
__global__ __launch_bounds__(256) void k_glu(
    const float* __restrict__ E, float* __restrict__ Y)
{
  int i = blockIdx.x * 256 + threadIdx.x;
  int c = blockIdx.y, b = blockIdx.z;
  const float* eb = E + (size_t)b * 128 * HWSZ;
  float a  = eb[(size_t)c * HWSZ + i];
  float g2 = eb[(size_t)(c + 64) * HWSZ + i];
  float ge = 0.5f * a * (1.f + erff(a * 0.70710678118f));
  Y[((size_t)b * CDIM + c) * HWSZ + i] = ge * g2;
}

// ===========================================================================
// Channel attention pieces.
// ===========================================================================
__global__ __launch_bounds__(256) void k_rownorm(
    const float* __restrict__ QK, float* __restrict__ Nrm)
{
  __shared__ float red[256];
  int bc = blockIdx.x;
  const float* p = QK + (size_t)bc * HWSZ;
  float s = 0.f;
  for (int i = threadIdx.x; i < HWSZ; i += 256) { float v = p[i]; s += v * v; }
  red[threadIdx.x] = s;
  __syncthreads();
  for (int st = 128; st > 0; st >>= 1) {
    if (threadIdx.x < st) red[threadIdx.x] += red[threadIdx.x + st];
    __syncthreads();
  }
  if (threadIdx.x == 0) Nrm[bc] = fmaxf(sqrtf(red[0]), 1e-12f);
}

__global__ __launch_bounds__(256) void k_chdot(
    const float* __restrict__ QK, const float* __restrict__ Nrm,
    const float* __restrict__ temp, float* __restrict__ Sc)
{
  __shared__ float red[256];
  int j = blockIdx.x, ii = blockIdx.y, b = blockIdx.z;
  const float* qp = QK + ((size_t)b * CDIM + ii) * HWSZ;
  const float* kp = QK + ((size_t)b * CDIM + 32 + j) * HWSZ;
  float s = 0.f;
  for (int i = threadIdx.x; i < HWSZ; i += 256) s += qp[i] * kp[i];
  red[threadIdx.x] = s;
  __syncthreads();
  for (int st = 128; st > 0; st >>= 1) {
    if (threadIdx.x < st) red[threadIdx.x] += red[threadIdx.x + st];
    __syncthreads();
  }
  if (threadIdx.x == 0)
    Sc[((size_t)b * 32 + ii) * 32 + j] =
        red[0] * temp[0] / (Nrm[b * CDIM + ii] * Nrm[b * CDIM + 32 + j]);
}

__global__ __launch_bounds__(64) void k_softmax32(float* __restrict__ Sc)
{
  float* p = Sc + threadIdx.x * 32;
  float mx = -1e30f;
  for (int j = 0; j < 32; ++j) mx = fmaxf(mx, p[j]);
  float sum = 0.f;
  for (int j = 0; j < 32; ++j) { p[j] = __expf(p[j] - mx); sum += p[j]; }
  float inv = 1.f / sum;
  for (int j = 0; j < 32; ++j) p[j] *= inv;
}

__global__ __launch_bounds__(256) void k_chav(
    const float* __restrict__ Sc, const float* __restrict__ VS,
    float* __restrict__ Y)
{
  __shared__ float A[32 * 32];
  int b = blockIdx.x >> 7, pt = blockIdx.x & 127;
  for (int t = threadIdx.x; t < 1024; t += 256) A[t] = Sc[b * 1024 + t];
  __syncthreads();
  int n = pt * 256 + threadIdx.x;
  float acc[32];
#pragma unroll
  for (int i = 0; i < 32; ++i) acc[i] = 0.f;
  for (int j = 0; j < 32; ++j) {
    float vv = VS[((size_t)b * CDIM + j) * HWSZ + n];
#pragma unroll
    for (int i = 0; i < 32; ++i) acc[i] += A[i * 32 + j] * vv;
  }
#pragma unroll
  for (int i = 0; i < 32; ++i)
    Y[((size_t)b * CDIM + i) * HWSZ + n] = acc[i];
}

// ===========================================================================
// Host orchestration
// ===========================================================================
extern "C" void kernel_launch(void* const* d_in, const int* in_sizes, int n_in,
                              void* d_out, int out_size, void* d_ws, size_t ws_size,
                              hipStream_t stream) {
  (void)in_sizes; (void)out_size; (void)ws_size;
  const size_t NBf = (size_t)BDIM * CDIM * HWSZ;  // 4,194,304 floats / tensor

  auto Wp = [&](int i) -> const float* {
    if (i >= n_in) i = n_in - 1;                   // safety clamp
    return (const float*)d_in[i];
  };

  float* WSF = (float*)d_ws;
  float* XL = WSF;                 float* XR = WSF + NBf;
  float* T0 = WSF + 2 * NBf;       float* T1 = WSF + 3 * NBf;
  float* T2 = WSF + 4 * NBf;       float* T3 = WSF + 5 * NBf;
  float* T4 = WSF + 6 * NBf;       float* T5 = WSF + 7 * NBf;
  float* EB  = WSF + 8 * NBf;                      // 2 units (128ch)
  float* EB2 = WSF + 10 * NBf;                     // reuses packed region
  _Float16* Pbase = (_Float16*)(WSF + 10 * NBf);
  auto Pk = [&](int k) { return Pbase + (size_t)k * NBf; };  // 8MB slots
  float* SM  = WSF + 13 * NBf;
  float* GT  = SM + 256;
  float* NRM = SM + 512;
  float* SC  = SM + 1024;
  _Float16* WH = (_Float16*)(SM + 4096);           // packed f16 weights

  hipMemcpyAsync(XL, d_in[0], NBf * sizeof(float), hipMemcpyDeviceToDevice, stream);
  hipMemcpyAsync(XR, d_in[1], NBf * sizeof(float), hipMemcpyDeviceToDevice, stream);

  auto conv = [&](const float* X, int wi, int bi, const float* res,
                  float* Y, int Cout) {
    int n = Cout * CDIM;
    k_packw<<<(n + 255) / 256, 256, 0, stream>>>(Wp(wi), WH, n);
    const float* bp = bi >= 0 ? Wp(bi) : nullptr;
    if (bp && res)
      k_conv1x1_lds<true, true><<<NPIX / 128, 256, 0, stream>>>(X, WH, bp, res, Y, Cout);
    else if (bp)
      k_conv1x1_lds<true, false><<<NPIX / 128, 256, 0, stream>>>(X, WH, bp, nullptr, Y, Cout);
    else if (res)
      k_conv1x1_lds<false, true><<<NPIX / 128, 256, 0, stream>>>(X, WH, nullptr, res, Y, Cout);
    else
      k_conv1x1_lds<false, false><<<NPIX / 128, 256, 0, stream>>>(X, WH, nullptr, nullptr, Y, Cout);
  };
  auto ln = [&](const float* X, int wi, int bi, float* Y) {
    k_ln2d<<<NPIX / 256, 256, 0, stream>>>(X, Wp(wi), Wp(bi), Y);
  };
  auto dw = [&](const float* X, int wi, int bi, float* Y, int C) {
    dim3 g(HWSZ / 256, C, BDIM);
    k_dw3x3<<<g, 256, 0, stream>>>(X, Wp(wi), bi >= 0 ? Wp(bi) : nullptr, Y, C);
  };
  auto fusion = [&](const float* x1, const float* x2, int f1i, int f2i, float* Y) {
    k_mean_hw<<<BDIM * CDIM, 256, 0, stream>>>(x1, x2, SM);
    k_se<<<BDIM, 64, 0, stream>>>(SM, Wp(f1i), Wp(f2i), GT);
    dim3 g(HWSZ / 256, CDIM, BDIM);
    k_gate<<<g, 256, 0, stream>>>(x1, x2, GT, Y, CDIM, 1);
  };

  // ---- self attention (dw_b+0 dw_w+1 f1+2 f2+3 ln_b+4 ln_w+5 proj_b+6
  //      proj_w+7 qk_b+8 qk_w+9 vs_b+10 vs_w+11)
  auto self_attn = [&](int p, float* X, int shift) {
    int sh = shift ? 8 : 0;
    ln(X, p + 5, p + 4, T0);
    conv(T0, p + 9, p + 8, nullptr, T1, CDIM);           // qk
    conv(T0, p + 11, p + 10, nullptr, T2, CDIM);         // vs
    dw(T2, p + 1, p + 0, T3, CDIM);                      // local
    k_pack_win<<<8192, 256, 0, stream>>>(T1, Pk(0), 0,  sh);   // q
    k_pack_win<<<8192, 256, 0, stream>>>(T1, Pk(1), 32, sh);   // k
    k_pack_win_vt<<<8192, 256, 0, stream>>>(T2, Pk(2), sh);    // v1 (transp)
    k_attn<16, 16><<<512, 128, 0, stream>>>(Pk(0), Pk(1), Pk(2), Pk(3), 0.25f);
    k_unpack_win<<<8192, 256, 0, stream>>>(Pk(3), T4, sh);     // y -> ch 0..31
    { dim3 g(HWSZ / 256, 32, BDIM);
      k_copy_hi<<<g, 256, 0, stream>>>(T2, T4); }              // v2 -> ch 32..63
    fusion(T4, T3, p + 2, p + 3, T5);
    conv(T5, p + 7, p + 6, X, X, CDIM);                  // proj + residual
  };

  // ---- channel attention (d1+0 d2+1 f1+2 f2+3 ln_b+4 ln_w+5 proj_b+6
  //      proj_w+7 qk1_b+8 qk1_w+9 qk2_b+10 qk2_w+11 temp+12 v1_b+13 v1_w+14
  //      v2_b+15 v2_w+16)
  auto channel_attn = [&](int p, float* X) {
    ln(X, p + 5, p + 4, T0);
    conv(T0, p + 9, p + 8, nullptr, T1, CDIM);
    dw(T1, p + 11, p + 10, T2, CDIM);                    // qk
    conv(T0, p + 14, p + 13, nullptr, T1, CDIM);
    dw(T1, p + 16, p + 15, T3, CDIM);                    // vs
    k_mean_hw<<<BDIM * CDIM, 256, 0, stream>>>(T3, nullptr, SM);
    k_se<<<BDIM, 64, 0, stream>>>(SM, Wp(p + 0), Wp(p + 1), GT);
    { dim3 g(HWSZ / 256, CDIM, BDIM);
      k_gate<<<g, 256, 0, stream>>>(T3, nullptr, GT, T4, CDIM, 0); }  // local
    k_rownorm<<<BDIM * CDIM, 256, 0, stream>>>(T2, NRM);
    { dim3 g(32, 32, BDIM);
      k_chdot<<<g, 256, 0, stream>>>(T2, NRM, Wp(p + 12), SC); }
    k_softmax32<<<1, 64, 0, stream>>>(SC);
    k_chav<<<BDIM * 128, 256, 0, stream>>>(SC, T3, T5);  // y -> ch 0..31
    { dim3 g(HWSZ / 256, 32, BDIM);
      k_copy_hi<<<g, 256, 0, stream>>>(T3, T5); }        // v2 -> ch 32..63
    fusion(T5, T4, p + 2, p + 3, T1);
    conv(T1, p + 7, p + 6, X, X, CDIM);
  };

  // ---- cross attention (feaL_w+0 feaR_w+1 lnL_b+2 lnL_w+3 lnR_b+4 lnR_w+5
  //      out_w+6 t1_b+7 t1_w+8 t2_b+9 t2_w+10 to_l_w+11 to_r_w+12)
  auto cross_attn = [&](int p) {
    conv(XL, p + 0, -1, nullptr, T0, CDIM);              // vl
    conv(XR, p + 1, -1, nullptr, T1, CDIM);              // vr
    dw(XL, p + 8, p + 7, T2, CDIM);
    conv(T2, p + 10, p + 9, XL, T3, CDIM);               // trans(xl)
    ln(T3, p + 3, p + 2, T2);
    conv(T2, p + 11, -1, nullptr, T4, 48);               // xl features
    dw(XR, p + 8, p + 7, T2, CDIM);
    conv(T2, p + 10, p + 9, XR, T5, CDIM);               // trans(xr)
    ln(T5, p + 5, p + 4, T2);
    conv(T2, p + 12, -1, nullptr, T5, 48);               // xr features
    k_pack_rows<<<BDIM * HDIM, 256, 0, stream>>>(T4, Pk(0), 48);
    k_pack_rows<<<BDIM * HDIM, 256, 0, stream>>>(T5, Pk(1), 48);
    k_pack_rows_vt<<<BDIM * HDIM, 256, 0, stream>>>(T1, Pk(2));   // vr^T
    k_pack_rows_vt<<<BDIM * HDIM, 256, 0, stream>>>(T0, Pk(3));   // vl^T
    k_attn<48, 64><<<BDIM * HDIM, 128, 0, stream>>>(Pk(0), Pk(1), Pk(2),
                                                    Pk(4), 1.0f); // warpL
    k_attn<48, 64><<<BDIM * HDIM, 128, 0, stream>>>(Pk(1), Pk(0), Pk(3),
                                                    Pk(5), 1.0f); // warpR
    k_unpack_rows<<<BDIM * HDIM, 256, 0, stream>>>(Pk(4), T2);
    conv(T2, p + 6, -1, XL, XL, CDIM);
    k_unpack_rows<<<BDIM * HDIM, 256, 0, stream>>>(Pk(5), T2);
    conv(T2, p + 6, -1, XR, XR, CDIM);
  };

  // ---- MLP (dw_b+0 dw_w+1 ln_b+2 ln_w+3 p1_b+4 p1_w+5 po_b+6 po_w+7)
  auto mlp = [&](int p, float* X) {
    ln(X, p + 3, p + 2, T0);
    conv(T0, p + 5, p + 4, nullptr, EB, 128);
    dw(EB, p + 1, p + 0, EB2, 128);
    { dim3 g(HWSZ / 256, CDIM, BDIM);
      k_glu<<<g, 256, 0, stream>>>(EB2, T1); }
    conv(T1, p + 7, p + 6, X, X, CDIM);
  };

  // -------- block schedule (pytree-flatten indices, sorted dict keys) -------
  cross_attn(2);                                   // blk0
  mlp(15, XL); mlp(15, XR);
  self_attn(31, XL, 0); self_attn(31, XR, 0);      // blk1
  mlp(23, XL); mlp(23, XR);
  channel_attn(43, XL); channel_attn(43, XR);      // blk2
  mlp(60, XL); mlp(60, XR);
  self_attn(76, XL, 1); self_attn(76, XR, 1);      // blk3 (shifted)
  mlp(68, XL); mlp(68, XR);
  channel_attn(88, XL); channel_attn(88, XR);      // blk4
  mlp(105, XL); mlp(105, XR);
  self_attn(121, XL, 0); self_attn(121, XR, 0);    // blk5
  mlp(113, XL); mlp(113, XR);
  float* outL = (float*)d_out;
  float* outR = outL + NBf;
  conv(XL, 134, 133, (const float*)d_in[0], outL, CDIM);
  conv(XR, 134, 133, (const float*)d_in[1], outR, CDIM);
}
